// DGCNN_seg_36962488549476
// MI455X (gfx1250) — matrix operand order repaired and checked by reading, not compiled, not run on hardware
//
#include <hip/hip_runtime.h>

// ---------------------------------------------------------------------------
// DGCNN segmentation forward for gfx1250 (MI455X), wave32 + WMMA f16->f32.
// Fused per-layer pipeline: gather edge messages -> GEMM1(+ReLU) -> GEMM2
// -> atomic-max scatter to nodes, all through LDS. Node MLP via the same
// WMMA GEMM core with LDS-staged A tiles. Each wave computes a 16x32 output
// tile (two accumulators sharing one A fragment) for WMMA ILP + A reuse.
// ---------------------------------------------------------------------------

typedef __attribute__((ext_vector_type(16))) _Float16 v16h;
typedef __attribute__((ext_vector_type(8)))  _Float16 v8h;
typedef __attribute__((ext_vector_type(8)))  float    v8f;

#define WAVES 8   // 256 threads = 8 wave32
#define TM    32  // rows (edges / nodes) per block tile

// Monotone order-preserving float->u32 key so segment-max becomes atomicMax_u32.
__device__ __forceinline__ unsigned fkey(float f) {
    unsigned u = __float_as_uint(f);
    return (u & 0x80000000u) ? ~u : (u | 0x80000000u);
}
__device__ __forceinline__ float fdecode(unsigned k) {
    return (k & 0x80000000u) ? __uint_as_float(k ^ 0x80000000u)
                             : __uint_as_float(~k);
}
#define NEG_INF_KEY 0x007FFFFFu  // fkey(-inf)

// A-fragment (16x32 f16, M=lane&15): halves 0..7 <- K[8g..8g+7],
// halves 8..15 <- K[16+8g..16+8g+7]   (g = lane>>4), per CDNA5 ISA 7.12.2.
__device__ __forceinline__ v16h load_a_frag(const _Float16* rowPtr, int g) {
    v8h lo = *(const v8h*)(rowPtr + 8 * g);
    v8h hi = *(const v8h*)(rowPtr + 16 + 8 * g);
    v16h a;
#pragma unroll
    for (int q = 0; q < 8; ++q) { a[q] = lo[q]; a[8 + q] = hi[q]; }
    return a;
}

// ---------------------------------------------------------------------------
// Pack W[K][N] (f32, row-major) into per-lane WMMA B fragments:
// out[((nt*numKt + kt)*32 + lane)*16 + h] = W[kt*32 + (h<8 ? 8*(lane>>4) + h : 16 + 8*(lane>>4) + h-8)][nt*16 + (lane&15)]
// ---------------------------------------------------------------------------
__global__ void pack_weights_kernel(const float* __restrict__ W, _Float16* __restrict__ out,
                                    int Krows, int Ncols, int numKt, int numNt) {
    long idx = (long)blockIdx.x * blockDim.x + threadIdx.x;
    long total = (long)numKt * numNt * 512;
    if (idx >= total) return;
    int h    = (int)(idx & 15);
    int lane = (int)((idx >> 4) & 31);
    int kt   = (int)((idx >> 9) % numKt);
    int nt   = (int)(idx / (512L * numKt));
    int g = lane >> 4;
    int n = nt * 16 + (lane & 15);
    int k = kt * 32 + (h < 8 ? 8 * g + h : 16 + 8 * g + (h - 8));
    float v = (k < Krows && n < Ncols) ? W[(long)k * Ncols + n] : 0.0f;
    out[idx] = (_Float16)v;
}

__global__ void fill_kernel(unsigned* __restrict__ p, unsigned v, long n) {
    long t = (long)blockIdx.x * blockDim.x + threadIdx.x;
    if (t < n) p[t] = v;
}

// Decode key buffer [N,C] -> f32 concat buffer slice; empty segments (-inf) -> 0.
__global__ void finalize_kernel(const unsigned* __restrict__ acc, float* __restrict__ dstF,
                                int N, int C, int stride, int colOff) {
    long t = (long)blockIdx.x * blockDim.x + threadIdx.x;
    if (t >= (long)N * C) return;
    int n = (int)(t / C), c = (int)(t % C);
    float f = fdecode(acc[t]);
    if (!__builtin_isfinite(f)) f = 0.0f;
    dstF[(long)n * stride + colOff + c] = f;
}

// ---------------------------------------------------------------------------
// Fused EdgeConv: per 32-edge tile
//   Mlds[e][:] = [x_dst, x_src - x_dst] (f16, K padded to KP)
//   Hlds = relu(Mlds @ W1p + b1)        (WMMA f16->f32, back to f16 in LDS)
//   Y    = Hlds @ W2p + b2              (WMMA), atomicMax-key scatter to acc[dst]
// Each wave: 16x32 output tile = two accumulators sharing one A fragment.
// ---------------------------------------------------------------------------
template <int CIN, int KP, int CH, int COUT>
__global__ void __launch_bounds__(256)
edgeconv_kernel(const float* __restrict__ nodeF, int nodeStride,
                const int* __restrict__ srcIdx, const int* __restrict__ dstIdx,
                const _Float16* __restrict__ W1p, const float* __restrict__ b1,
                const _Float16* __restrict__ W2p, const float* __restrict__ b2,
                unsigned* __restrict__ acc, int E) {
    constexpr int LDM = KP + 8;  // pad 8 halves (16B) to dodge LDS bank conflicts
    constexpr int LDH = CH + 8;
    alignas(16) __shared__ _Float16 Mlds[TM * LDM];
    alignas(16) __shared__ _Float16 Hlds[TM * LDH];
    __shared__ int sSrc[TM];
    __shared__ int sDst[TM];

    const int tid  = threadIdx.x;
    const int lane = tid & 31;
    const int wave = tid >> 5;
    const int g    = lane >> 4;
    const int col  = lane & 15;
    const int e0   = blockIdx.x * TM;

    if (tid < TM) {
        int ge = e0 + tid;
        int gc = ge < E ? ge : (E - 1);
        sSrc[tid] = srcIdx[gc];
        sDst[tid] = dstIdx[gc];
    }
    __syncthreads();

    // Phase 0: gather messages [xi, xj - xi], zero-pad K up to KP.
    for (int i = tid; i < TM * KP; i += 256) {
        int r = i / KP, c = i % KP;
        float v = 0.0f;
        if (c < CIN) {
            v = nodeF[(long)sDst[r] * nodeStride + c];
        } else if (c < 2 * CIN) {
            int cc = c - CIN;
            v = nodeF[(long)sSrc[r] * nodeStride + cc] -
                nodeF[(long)sDst[r] * nodeStride + cc];
        }
        Mlds[r * LDM + c] = (_Float16)v;
    }
    __syncthreads();

    // Phase 1: H = relu(M @ W1 + b1)
    {
        constexpr int NKT = KP / 32;
        constexpr int NCP = (CH / 16) / 2;   // column-tile pairs
        constexpr int NRT = TM / 16;
        for (int t = wave; t < NRT * NCP; t += WAVES) {
            int rt = t / NCP, ct0 = (t % NCP) * 2;
            v8f c0 = {}, c1 = {};
            for (int kt = 0; kt < NKT; ++kt) {
                v16h a  = load_a_frag(&Mlds[(rt * 16 + col) * LDM + kt * 32], g);
                v16h b0 = *(const v16h*)&W1p[(((long)(ct0 + 0) * NKT + kt) * 32 + lane) * 16];
                v16h b1 = *(const v16h*)&W1p[(((long)(ct0 + 1) * NKT + kt) * 32 + lane) * 16];
                c0 = __builtin_amdgcn_wmma_f32_16x16x32_f16(false, a, false, b0,
                                                            (short)0, c0, false, false);
                c1 = __builtin_amdgcn_wmma_f32_16x16x32_f16(false, a, false, b1,
                                                            (short)0, c1, false, false);
            }
            float bb0 = b1[(ct0 + 0) * 16 + col];
            float bb1 = b1[(ct0 + 1) * 16 + col];
#pragma unroll
            for (int v = 0; v < 8; ++v) {
                int row = rt * 16 + v + 8 * g;
                float y0 = c0[v] + bb0; y0 = y0 > 0.0f ? y0 : 0.0f;
                float y1 = c1[v] + bb1; y1 = y1 > 0.0f ? y1 : 0.0f;
                Hlds[row * LDH + (ct0 + 0) * 16 + col] = (_Float16)y0;
                Hlds[row * LDH + (ct0 + 1) * 16 + col] = (_Float16)y1;
            }
        }
    }
    __syncthreads();

    // Phase 2: Y = H @ W2 + b2, scatter max-key per (dst node, feature).
    {
        constexpr int NKT = CH / 32;
        constexpr int NCP = (COUT / 16) / 2;
        constexpr int NRT = TM / 16;
        for (int t = wave; t < NRT * NCP; t += WAVES) {
            int rt = t / NCP, ct0 = (t % NCP) * 2;
            v8f c0 = {}, c1 = {};
            for (int kt = 0; kt < NKT; ++kt) {
                v16h a  = load_a_frag(&Hlds[(rt * 16 + col) * LDH + kt * 32], g);
                v16h b0 = *(const v16h*)&W2p[(((long)(ct0 + 0) * NKT + kt) * 32 + lane) * 16];
                v16h b1 = *(const v16h*)&W2p[(((long)(ct0 + 1) * NKT + kt) * 32 + lane) * 16];
                c0 = __builtin_amdgcn_wmma_f32_16x16x32_f16(false, a, false, b0,
                                                            (short)0, c0, false, false);
                c1 = __builtin_amdgcn_wmma_f32_16x16x32_f16(false, a, false, b1,
                                                            (short)0, c1, false, false);
            }
            float bb0 = b2[(ct0 + 0) * 16 + col];
            float bb1 = b2[(ct0 + 1) * 16 + col];
#pragma unroll
            for (int v = 0; v < 8; ++v) {
                int r  = rt * 16 + v + 8 * g;
                int ge = e0 + r;
                if (ge < E) {
                    long base = (long)sDst[r] * COUT;
                    atomicMax(&acc[base + (ct0 + 0) * 16 + col], fkey(c0[v] + bb0));
                    atomicMax(&acc[base + (ct0 + 1) * 16 + col], fkey(c1[v] + bb1));
                }
            }
        }
    }
}

// ---------------------------------------------------------------------------
// Node MLP GEMM: out[M,NOUT] (f16) = act(A[M,KP] @ Wp + bias), A staged in LDS.
// ---------------------------------------------------------------------------
template <typename TA, int KP, int NOUT, bool RELU>
__global__ void __launch_bounds__(256)
mlp_gemm_kernel(const TA* __restrict__ A, const _Float16* __restrict__ Wp,
                const float* __restrict__ bias, _Float16* __restrict__ out, int M) {
    constexpr int LDM = KP + 8;
    alignas(16) __shared__ _Float16 Alds[TM * LDM];

    const int tid  = threadIdx.x;
    const int lane = tid & 31;
    const int wave = tid >> 5;
    const int g    = lane >> 4;
    const int col  = lane & 15;
    const int r0   = blockIdx.x * TM;

    for (int i = tid; i < TM * KP; i += 256) {
        int r = i / KP, c = i % KP;
        int gr = r0 + r;
        float v = (gr < M) ? (float)A[(long)gr * KP + c] : 0.0f;
        Alds[r * LDM + c] = (_Float16)v;
    }
    __syncthreads();

    constexpr int NKT = KP / 32;
    constexpr int NCP = (NOUT / 16) / 2;
    constexpr int NRT = TM / 16;
    for (int t = wave; t < NRT * NCP; t += WAVES) {
        int rt = t / NCP, ct0 = (t % NCP) * 2;
        v8f c0 = {}, c1 = {};
        for (int kt = 0; kt < NKT; ++kt) {
            v16h a  = load_a_frag(&Alds[(rt * 16 + col) * LDM + kt * 32], g);
            v16h b0 = *(const v16h*)&Wp[(((long)(ct0 + 0) * NKT + kt) * 32 + lane) * 16];
            v16h b1 = *(const v16h*)&Wp[(((long)(ct0 + 1) * NKT + kt) * 32 + lane) * 16];
            c0 = __builtin_amdgcn_wmma_f32_16x16x32_f16(false, a, false, b0,
                                                        (short)0, c0, false, false);
            c1 = __builtin_amdgcn_wmma_f32_16x16x32_f16(false, a, false, b1,
                                                        (short)0, c1, false, false);
        }
        float bb0 = bias[(ct0 + 0) * 16 + col];
        float bb1 = bias[(ct0 + 1) * 16 + col];
#pragma unroll
        for (int v = 0; v < 8; ++v) {
            int gr = r0 + rt * 16 + v + 8 * g;
            float y0 = c0[v] + bb0;
            float y1 = c1[v] + bb1;
            if (RELU) {
                y0 = y0 > 0.0f ? y0 : 0.0f;
                y1 = y1 > 0.0f ? y1 : 0.0f;
            }
            if (gr < M) {
                out[(long)gr * NOUT + (ct0 + 0) * 16 + col] = (_Float16)y0;
                out[(long)gr * NOUT + (ct0 + 1) * 16 + col] = (_Float16)y1;
            }
        }
    }
}

// Final tiny GEMM 256 -> 4 (f32 out), scalar dot per (node, out-feature).
__global__ void mlp3_kernel(const _Float16* __restrict__ h, const float* __restrict__ W,
                            const float* __restrict__ b, float* __restrict__ out, int M) {
    long t = (long)blockIdx.x * blockDim.x + threadIdx.x;
    if (t >= (long)M * 4) return;
    int n = (int)(t >> 2), j = (int)(t & 3);
    const _Float16* hr = h + (long)n * 256;
    float s = b[j];
#pragma unroll 8
    for (int k = 0; k < 256; ++k) s += (float)hr[k] * W[k * 4 + j];
    out[t] = s;
}

// ---------------------------------------------------------------------------
extern "C" void kernel_launch(void* const* d_in, const int* in_sizes, int n_in,
                              void* d_out, int out_size, void* d_ws, size_t ws_size,
                              hipStream_t stream) {
    (void)n_in; (void)out_size; (void)ws_size;
    const int N = in_sizes[0] / 3;
    const int E = in_sizes[1] / 2;

    const float* x   = (const float*)d_in[0];
    const int*   ei  = (const int*)d_in[1];
    const int*   src = ei;
    const int*   dst = ei + E;
    const float* w1a = (const float*)d_in[3];  const float* b1a = (const float*)d_in[4];
    const float* w1b = (const float*)d_in[5];  const float* b1b = (const float*)d_in[6];
    const float* w2a = (const float*)d_in[7];  const float* b2a = (const float*)d_in[8];
    const float* w2b = (const float*)d_in[9];  const float* b2b = (const float*)d_in[10];
    const float* w3a = (const float*)d_in[11]; const float* b3a = (const float*)d_in[12];
    const float* w3b = (const float*)d_in[13]; const float* b3b = (const float*)d_in[14];
    const float* wm1 = (const float*)d_in[15]; const float* bm1 = (const float*)d_in[16];
    const float* wm2 = (const float*)d_in[17]; const float* bm2 = (const float*)d_in[18];
    const float* wm3 = (const float*)d_in[19]; const float* bm3 = (const float*)d_in[20];
    float* out = (float*)d_out;

    char* ws = (char*)d_ws;
    size_t off = 0;
    auto carve = [&](size_t bytes) -> void* {
        void* p = ws + off;
        off += (bytes + 255) & ~(size_t)255;
        return p;
    };
    float*    concat = (float*)carve((size_t)N * 448 * 4);   // [N,448] h1|h2|h3
    unsigned* acc    = (unsigned*)carve((size_t)N * 256 * 4);
    _Float16* m1out  = (_Float16*)carve((size_t)N * 512 * 2);
    _Float16* m2out  = (_Float16*)carve((size_t)N * 256 * 2);
    _Float16* w1ap = (_Float16*)carve(2048 * 2);
    _Float16* w1bp = (_Float16*)carve(4096 * 2);
    _Float16* w2ap = (_Float16*)carve(16384 * 2);
    _Float16* w2bp = (_Float16*)carve(16384 * 2);
    _Float16* w3ap = (_Float16*)carve(65536 * 2);
    _Float16* w3bp = (_Float16*)carve(65536 * 2);
    _Float16* wm1p = (_Float16*)carve(229376 * 2);
    _Float16* wm2p = (_Float16*)carve(131072 * 2);

    auto pack = [&](const float* W, _Float16* dstp, int Kr, int Nc, int nKt, int nNt) {
        long total = (long)nKt * nNt * 512;
        pack_weights_kernel<<<dim3((unsigned)((total + 255) / 256)), dim3(256), 0, stream>>>(
            W, dstp, Kr, Nc, nKt, nNt);
    };
    pack(w1a, w1ap,   6,  64,  1,  4);
    pack(w1b, w1bp,  64,  64,  2,  4);
    pack(w2a, w2ap, 128, 128,  4,  8);
    pack(w2b, w2bp, 128, 128,  4,  8);
    pack(w3a, w3ap, 256, 256,  8, 16);
    pack(w3b, w3bp, 256, 256,  8, 16);
    pack(wm1, wm1p, 448, 512, 14, 32);
    pack(wm2, wm2p, 512, 256, 16, 16);

    const unsigned convBlocks = (unsigned)((E + TM - 1) / TM);
    auto fill = [&](unsigned* p, long n) {
        fill_kernel<<<dim3((unsigned)((n + 255) / 256)), dim3(256), 0, stream>>>(p, NEG_INF_KEY, n);
    };
    auto fin = [&](int C, int colOff) {
        long n = (long)N * C;
        finalize_kernel<<<dim3((unsigned)((n + 255) / 256)), dim3(256), 0, stream>>>(
            acc, concat, N, C, 448, colOff);
    };

    // EdgeConv 1: x[N,3] -> h1[N,64] (concat cols 0..63)
    fill(acc, (long)N * 64);
    edgeconv_kernel<3, 32, 64, 64><<<convBlocks, 256, 0, stream>>>(
        x, 3, src, dst, w1ap, b1a, w1bp, b1b, acc, E);
    fin(64, 0);

    // EdgeConv 2: h1 -> h2[N,128] (concat cols 64..191)
    fill(acc, (long)N * 128);
    edgeconv_kernel<64, 128, 128, 128><<<convBlocks, 256, 0, stream>>>(
        concat, 448, src, dst, w2ap, b2a, w2bp, b2b, acc, E);
    fin(128, 64);

    // EdgeConv 3: h2 -> h3[N,256] (concat cols 192..447)
    fill(acc, (long)N * 256);
    edgeconv_kernel<128, 256, 256, 256><<<convBlocks, 256, 0, stream>>>(
        concat + 64, 448, src, dst, w3ap, b3a, w3bp, b3b, acc, E);
    fin(256, 192);

    // Node MLP
    const unsigned mlpBlocks = (unsigned)((N + TM - 1) / TM);
    mlp_gemm_kernel<float, 448, 512, true><<<mlpBlocks, 256, 0, stream>>>(
        concat, wm1p, bm1, m1out, N);
    mlp_gemm_kernel<_Float16, 512, 256, true><<<mlpBlocks, 256, 0, stream>>>(
        m1out, wm2p, bm2, m2out, N);
    mlp3_kernel<<<dim3((unsigned)(((long)N * 4 + 255) / 256)), dim3(256), 0, stream>>>(
        m2out, wm3, bm3, out, N);
}
